// MultiHeadAttention_56384330662532
// MI455X (gfx1250) — matrix operand order repaired and checked by reading, not compiled, not run on hardware
//
#include <hip/hip_runtime.h>
#include <hip/hip_bf16.h>

typedef _Float16 v16h __attribute__((ext_vector_type(16)));
typedef _Float16 v8h  __attribute__((ext_vector_type(8)));
typedef _Float16 v4h  __attribute__((ext_vector_type(4)));
typedef float    v8f  __attribute__((ext_vector_type(8)));

#define BB 4
#define SS 2048
#define DD 1024
#define HH 16
#define HDIM 64

__device__ __forceinline__ v16h combine8(v8h lo, v8h hi) {
  return __builtin_shufflevector(lo, hi, 0,1,2,3,4,5,6,7,8,9,10,11,12,13,14,15);
}

__device__ __forceinline__ v4h cvt4(float4 f) {
  v4h r; r[0]=(_Float16)f.x; r[1]=(_Float16)f.y; r[2]=(_Float16)f.z; r[3]=(_Float16)f.w;
  return r;
}

__device__ __forceinline__ v8f wmma_f16(v16h a, v16h b, v8f c) {
  return __builtin_amdgcn_wmma_f32_16x16x32_f16(false, a, false, b, (short)0, c, false, false);
}

// async global(f16)->LDS 16-byte copy; tracked by ASYNCcnt
__device__ __forceinline__ void async_b128(uint32_t lds_off, const void* gaddr) {
  asm volatile("global_load_async_to_lds_b128 %0, %1, off"
               :: "v"(lds_off), "v"(gaddr) : "memory");
}
__device__ __forceinline__ void wait_async0() {
  asm volatile("s_wait_asynccnt 0x0" ::: "memory");
}
__device__ __forceinline__ uint32_t lds_off32(const void* p) {
  return (uint32_t)(size_t)p;
}

// ---------------------------------------------------------------------------
// C[M=8192,N=1024] = X[M,K=1024] @ W[N,K]^T + bias
//   XHALF=false: X is f32 (converted to f16 on the fly)
//   XHALF=true : X is f16 (A tile copied via async global->LDS)
// outA : f32 [M,N]            (optional)
// outB : f32 [B,H,S,64]       (optional)
// outAh: f16 [M,N]            (optional)
// outBh: f16 [B,H,S,64]       (optional)
// ---------------------------------------------------------------------------
template <bool XHALF>
__global__ __launch_bounds__(256)
void gemm_xwt(const void* __restrict__ Xv, const float* __restrict__ W,
              const float* __restrict__ bias,
              float* __restrict__ outA, float* __restrict__ outB,
              _Float16* __restrict__ outAh, _Float16* __restrict__ outBh) {
  const int K = DD, N = DD;
  __shared__ _Float16 As[128 * 48];   // 128 rows x 32 halves, stride 48 (16B-aligned rows)
  __shared__ _Float16 Bs[128 * 48];

  const int t    = threadIdx.x;
  const int lane = t & 31;
  const int w    = t >> 5;
  const int wm   = w & 3;        // 4 wave-rows  * 32 rows
  const int wn   = w >> 2;       // 2 wave-cols  * 64 cols
  const int m0   = blockIdx.y * 128;
  const int n0   = blockIdx.x * 128;

  v8f acc[2][4];
  #pragma unroll
  for (int i = 0; i < 2; ++i)
    #pragma unroll
    for (int j = 0; j < 4; ++j) acc[i][j] = (v8f){};

  const int nl  = lane & 15;
  const int alo = (lane < 16) ? 0 : 8;    // A-frag k pattern {alo..+7, alo+16..+23}
  const int bko = (lane < 16) ? 0 : 8;    // B-frag k pattern {bko..+7, bko+16..+23}

  for (int k0 = 0; k0 < K; k0 += 32) {
    __syncthreads();
    // ---- A tile: 128x32 halves ----
    if (XHALF) {
      const _Float16* X16 = (const _Float16*)Xv;
      #pragma unroll
      for (int i = 0; i < 2; ++i) {
        int ch = t + i * 256;              // 0..511 16B chunks
        int r  = ch >> 2, c8 = ch & 3;     // 4 chunks per row
        async_b128(lds_off32(As + r * 48 + c8 * 8),
                   X16 + (size_t)(m0 + r) * K + k0 + c8 * 8);
      }
    } else {
      const float* X32 = (const float*)Xv;
      #pragma unroll
      for (int i = 0; i < 4; ++i) {
        int f4 = t + i * 256;              // 0..1023 float4 per tile
        int r  = f4 >> 3, c4 = f4 & 7;
        float4 va = *(const float4*)(X32 + (size_t)(m0 + r) * K + k0 + c4 * 4);
        *(v4h*)(As + r * 48 + c4 * 4) = cvt4(va);
      }
    }
    // ---- B tile: W f32 -> f16 ----
    #pragma unroll
    for (int i = 0; i < 4; ++i) {
      int f4 = t + i * 256;
      int r  = f4 >> 3, c4 = f4 & 7;
      float4 vb = *(const float4*)(W + (size_t)(n0 + r) * K + k0 + c4 * 4);
      *(v4h*)(Bs + r * 48 + c4 * 4) = cvt4(vb);
    }
    if (XHALF) wait_async0();
    __syncthreads();

    v16h afrag[2];
    #pragma unroll
    for (int tm = 0; tm < 2; ++tm) {
      int r = wm * 32 + tm * 16 + nl;
      afrag[tm] = combine8(*(const v8h*)(As + r * 48 + alo),
                           *(const v8h*)(As + r * 48 + alo + 16));
    }
    #pragma unroll
    for (int tn = 0; tn < 4; ++tn) {
      int r = wn * 64 + tn * 16 + nl;
      v16h bfrag = combine8(*(const v8h*)(Bs + r * 48 + bko),
                            *(const v8h*)(Bs + r * 48 + bko + 16));
      acc[0][tn] = wmma_f16(afrag[0], bfrag, acc[0][tn]);
      acc[1][tn] = wmma_f16(afrag[1], bfrag, acc[1][tn]);
    }
  }

  // epilogue: C layout -> lane nl holds column, VGPR r holds row (+8 upper half-lanes)
  const int rh = (lane >> 4) * 8;
  #pragma unroll
  for (int tm = 0; tm < 2; ++tm) {
    #pragma unroll
    for (int tn = 0; tn < 4; ++tn) {
      int n = n0 + wn * 64 + tn * 16 + nl;
      float bv = bias[n];
      #pragma unroll
      for (int r = 0; r < 8; ++r) {
        int m = m0 + wm * 32 + tm * 16 + rh + r;
        float vv = acc[tm][tn][r] + bv;
        if (outA)  outA[(size_t)m * N + n] = vv;
        if (outAh) outAh[(size_t)m * N + n] = (_Float16)vv;
        if (outB || outBh) {
          int b = m >> 11, s = m & (SS - 1);
          int h = n >> 6,  hd = n & 63;
          size_t idx = (((size_t)(b * HH + h) * SS) + s) * HDIM + hd;
          if (outB)  outB[idx] = vv;
          if (outBh) outBh[idx] = (_Float16)vv;
        }
      }
    }
  }
}

// ---------------------------------------------------------------------------
// Flash attention, all-f16 inputs: qp_h [B,S,D], kh_h/vh_h [B,H,S,64].
// Scores scaled by 0.125 in f32 after WMMA (exact), causal mask = +(-1e5) for
// j>i, matching the reference numerics. Output ao_h f16 [B,S,D].
// K tile loaded via async global->LDS; V tile manually transposed into LDS.
// ---------------------------------------------------------------------------
__global__ __launch_bounds__(256)
void attn_fwd(const _Float16* __restrict__ qp_h, const _Float16* __restrict__ kh_h,
              const _Float16* __restrict__ vh_h, _Float16* __restrict__ ao_h) {
  __shared__ _Float16 Kt[64 * 72];        // [key][d]  stride 72
  __shared__ _Float16 Vt[64 * 72];        // [d][key]  (transposed), stride 72
  __shared__ _Float16 Pb[8 * 16 * 72];    // per-wave P tile, 16 x 64, stride 72

  const int t    = threadIdx.x;
  const int lane = t & 31;
  const int w    = t >> 5;
  const int bh   = blockIdx.y;            // b*H + h
  const int b    = bh >> 4;
  const int h    = bh & 15;
  const int q0   = blockIdx.x * 128;
  const int qb   = q0 + w * 16;           // this wave's 16 query rows
  const int nl   = lane & 15;
  const int rh   = (lane >> 4) * 8;

  // ---- Q fragment (16 x 64) straight f16 copy, kept in registers ----
  v16h qf[2];
  {
    const _Float16* qr = qp_h + (size_t)(b * SS + qb + nl) * DD + h * HDIM;
    #pragma unroll
    for (int c = 0; c < 2; ++c) {
      int klo = c * 32 + ((lane < 16) ? 0 : 8);
      qf[c] = combine8(*(const v8h*)(qr + klo), *(const v8h*)(qr + klo + 16));
    }
  }

  float mprev[8], lsum[8];
  v8f oacc[4];
  #pragma unroll
  for (int r = 0; r < 8; ++r) { mprev[r] = -3.0e38f; lsum[r] = 0.0f; }
  #pragma unroll
  for (int i = 0; i < 4; ++i) oacc[i] = (v8f){};

  const int bko = (lane < 16) ? 0 : 8;
  const int nkt = 2 * blockIdx.x + 2;     // causal: only tiles with keys <= q0+127

  for (int kt = 0; kt < nkt; ++kt) {
    const int kb = kt * 64;
    __syncthreads();
    // ---- K tile: async f16 global -> LDS (row-major, padded stride) ----
    const _Float16* Kg = kh_h + ((size_t)bh * SS + kb) * HDIM;
    const _Float16* Vg = vh_h + ((size_t)bh * SS + kb) * HDIM;
    #pragma unroll
    for (int i = 0; i < 2; ++i) {
      int ch = t + i * 256;               // 0..511 16B chunks (64 rows x 8)
      int r  = ch >> 3, c8 = ch & 7;
      async_b128(lds_off32(Kt + r * 72 + c8 * 8), Kg + r * HDIM + c8 * 8);
      // V tile: load 8 halves of a row, store transposed
      v8h vv = *(const v8h*)(Vg + r * HDIM + c8 * 8);
      #pragma unroll
      for (int e = 0; e < 8; ++e) Vt[(c8 * 8 + e) * 72 + r] = vv[e];
    }
    wait_async0();
    __syncthreads();

    // ---- S = Q @ K^T for 64 keys (4 n-tiles x 2 k-chunks) ----
    v8f sacc[4];
    #pragma unroll
    for (int tn = 0; tn < 4; ++tn) {
      int kr = tn * 16 + nl;
      v16h bf0 = combine8(*(const v8h*)(Kt + kr * 72 + bko),
                          *(const v8h*)(Kt + kr * 72 + bko + 16));
      v16h bf1 = combine8(*(const v8h*)(Kt + kr * 72 + 32 + bko),
                          *(const v8h*)(Kt + kr * 72 + 32 + bko + 16));
      v8f z = (v8f){};
      z = wmma_f16(qf[0], bf0, z);
      sacc[tn] = wmma_f16(qf[1], bf1, z);
    }

    // ---- scale (1/sqrt(64), exact) + causal mask + per-row max ----
    float rmax[8];
    #pragma unroll
    for (int r = 0; r < 8; ++r) rmax[r] = -3.0e38f;
    #pragma unroll
    for (int tn = 0; tn < 4; ++tn) {
      int j = kb + tn * 16 + nl;
      #pragma unroll
      for (int r = 0; r < 8; ++r) {
        int qi = qb + rh + r;
        float sv = sacc[tn][r] * 0.125f;
        if (j > qi) sv += -100000.0f;
        sacc[tn][r] = sv;
        rmax[r] = fmaxf(rmax[r], sv);
      }
    }
    #pragma unroll
    for (int r = 0; r < 8; ++r) {
      float v = rmax[r];
      v = fmaxf(v, __shfl_xor(v, 1, 32));
      v = fmaxf(v, __shfl_xor(v, 2, 32));
      v = fmaxf(v, __shfl_xor(v, 4, 32));
      v = fmaxf(v, __shfl_xor(v, 8, 32));
      float mnew = fmaxf(mprev[r], v);
      float alpha = __expf(mprev[r] - mnew);
      mprev[r] = mnew;
      lsum[r] *= alpha;
      #pragma unroll
      for (int d = 0; d < 4; ++d) oacc[d][r] *= alpha;
    }

    // ---- P = exp(S - m), row-sum, stash P (f16) in per-wave LDS ----
    float rsum[8];
    #pragma unroll
    for (int r = 0; r < 8; ++r) rsum[r] = 0.0f;
    _Float16* Pw = Pb + w * 16 * 72;
    #pragma unroll
    for (int tn = 0; tn < 4; ++tn) {
      #pragma unroll
      for (int r = 0; r < 8; ++r) {
        float p = __expf(sacc[tn][r] - mprev[r]);
        rsum[r] += p;
        Pw[(rh + r) * 72 + tn * 16 + nl] = (_Float16)p;
      }
    }
    #pragma unroll
    for (int r = 0; r < 8; ++r) {
      float v = rsum[r];
      v += __shfl_xor(v, 1, 32);
      v += __shfl_xor(v, 2, 32);
      v += __shfl_xor(v, 4, 32);
      v += __shfl_xor(v, 8, 32);
      lsum[r] += v;
    }

    // ---- O += P @ V (A-frag from LDS P, B-frag from transposed V) ----
    #pragma unroll
    for (int c = 0; c < 2; ++c) {
      int base = c * 32 + ((lane < 16) ? 0 : 8);
      v16h af = combine8(*(const v8h*)(Pw + nl * 72 + base),
                         *(const v8h*)(Pw + nl * 72 + base + 16));
      int kk = c * 32 + bko;
      #pragma unroll
      for (int d = 0; d < 4; ++d) {
        int drow = d * 16 + nl;
        v16h bf = combine8(*(const v8h*)(Vt + drow * 72 + kk),
                           *(const v8h*)(Vt + drow * 72 + kk + 16));
        oacc[d] = wmma_f16(af, bf, oacc[d]);
      }
    }
  }

  // ---- finalize: O / l, f16 write to ao_h in [B,S,D] layout ----
  #pragma unroll
  for (int r = 0; r < 8; ++r) lsum[r] = 1.0f / lsum[r];
  #pragma unroll
  for (int d = 0; d < 4; ++d) {
    #pragma unroll
    for (int r = 0; r < 8; ++r) {
      int qi = qb + rh + r;
      int dc = h * HDIM + d * 16 + nl;
      ao_h[(size_t)(b * SS + qi) * DD + dc] = (_Float16)(oacc[d][r] * lsum[r]);
    }
  }
}

// ---------------------------------------------------------------------------
extern "C" void kernel_launch(void* const* d_in, const int* in_sizes, int n_in,
                              void* d_out, int out_size, void* d_ws, size_t ws_size,
                              hipStream_t stream) {
  const float* q  = (const float*)d_in[0];
  const float* k  = (const float*)d_in[1];
  const float* v  = (const float*)d_in[2];
  // d_in[3] = mask (causal triu, implemented analytically)
  const float* Wq = (const float*)d_in[4];
  const float* bq = (const float*)d_in[5];
  const float* Wk = (const float*)d_in[6];
  const float* bk = (const float*)d_in[7];
  const float* Wv = (const float*)d_in[8];
  const float* bv = (const float*)d_in[9];
  const float* Wo = (const float*)d_in[10];
  const float* bo = (const float*)d_in[11];

  float* out = (float*)d_out;
  const size_t BSD = (size_t)BB * SS * DD;     // 8,388,608
  float* kh = out + BSD;                       // output #2: [B,H,S,64] f32
  float* vh = out + 2 * BSD;                   // output #3: [B,H,S,64] f32

  _Float16* wsh  = (_Float16*)d_ws;            // 64 MB of f16 scratch
  _Float16* qp_h = wsh;                        // [B,S,D]
  _Float16* kh_h = wsh + BSD;                  // [B,H,S,64]
  _Float16* vh_h = wsh + 2 * BSD;              // [B,H,S,64]
  _Float16* ao_h = wsh + 3 * BSD;              // [B,S,D]

  dim3 gg(DD / 128, (BB * SS) / 128);          // 8 x 64 blocks
  gemm_xwt<false><<<gg, 256, 0, stream>>>(q, Wq, bq, nullptr, nullptr, qp_h, nullptr);
  gemm_xwt<false><<<gg, 256, 0, stream>>>(k, Wk, bk, nullptr, kh, nullptr, kh_h);
  gemm_xwt<false><<<gg, 256, 0, stream>>>(v, Wv, bv, nullptr, vh, nullptr, vh_h);

  dim3 ga(SS / 128, BB * HH);                  // 16 x 64 blocks
  attn_fwd<<<ga, 256, 0, stream>>>(qp_h, kh_h, vh_h, ao_h);

  gemm_xwt<true><<<gg, 256, 0, stream>>>(ao_h, Wo, bo, out, nullptr, nullptr, nullptr);
}
